// GNN_15710990368947
// MI455X (gfx1250) — matrix operand order repaired and checked by reading, not compiled, not run on hardware
//
#include <hip/hip_runtime.h>
#include <hip/hip_bf16.h>

#define PNODES 20
#define TPB 128
#define HLD 320          // f16 row stride of node-feature LDS buffers (max K-pad 320)
#define KNN 5

typedef __attribute__((ext_vector_type(16))) _Float16 v16h;
typedef __attribute__((ext_vector_type(8)))  _Float16 v8h;
typedef __attribute__((ext_vector_type(4)))  _Float16 v4h;
typedef __attribute__((ext_vector_type(8)))  float    v8f;
typedef __attribute__((ext_vector_type(4)))  float    v4f;

// ---------------------------------------------------------------- helpers ---

__device__ __forceinline__ v8f zero8() {
    v8f z;
#pragma unroll
    for (int i = 0; i < 8; ++i) z[i] = 0.0f;
    return z;
}

__device__ __forceinline__ v8f wmma32(v16h a, v16h b, v8f c) {
    return __builtin_amdgcn_wmma_f32_16x16x32_f16(false, a, false, b, (short)0, c, false, false);
}

// A fragment (16x32 f16) from row-major f16 tile; rowptr = src + m*ld (m = row_base + lane&15).
// A map: elems [0..15] = contiguous K kb+16*half .. +15
__device__ __forceinline__ v16h loadA16(const _Float16* rowptr, int kb, int half) {
    v8h c0 = *(const v8h*)(rowptr + kb + 16 * half);
    v8h c1 = *(const v8h*)(rowptr + kb + 16 * half + 8);
    v16h a;
#pragma unroll
    for (int i = 0; i < 8; ++i) { a[i] = c0[i]; a[8 + i] = c1[i]; }
    return a;
}

// B fragment (32x16 f16) where source holds the TRANSPOSED tile row-major
// (rowptr = srcT + n*ld, n = n_base + lane&15).
// B map (this code's order, same as A): elems [0..15] = contiguous K kb+16*half .. +15
__device__ __forceinline__ v16h loadB16(const _Float16* rowptr, int kb, int half) {
    v8h c0 = *(const v8h*)(rowptr + kb + 16 * half);
    v8h c1 = *(const v8h*)(rowptr + kb + 16 * half + 8);
    v16h b;
#pragma unroll
    for (int i = 0; i < 8; ++i) { b[i] = c0[i]; b[8 + i] = c1[i]; }
    return b;
}

// ------------------------------------------------------------ GCN layer -----
// Hin (f16, 32 x HLD, zero-padded) -> Hout (same shape).
// GM: 32x32 f32 scratch; Gram matrix lives here first, then (as f16) the
// aggregation matrix M (G is dead once top-K is done).
__device__ __forceinline__ void gcn_layer(
        _Float16* Hin, _Float16* Hout, _Float16* tT,
        float* GM, _Float16* Cm, float* dinvS,
        const _Float16* __restrict__ Wf,   // (nTiles*16) x KpadW, f16, zero-padded
        const float* __restrict__ bvec,
        int kTiles, int nTiles, int N_out, int KpadW,
        int tid, int lane, int wave) {
    __builtin_assume(Hin != nullptr);
    __builtin_assume(Hout != nullptr);
    __builtin_assume(tT != nullptr);
    __builtin_assume(GM != nullptr);
    __builtin_assume(Cm != nullptr);

    int lm   = lane & 15;
    int half = (lane >> 4) & 1;
    _Float16* Mm = (_Float16*)GM;          // aliases first 2 KB of GM

    // ---- zero Hout + Cm
    {
        uint4 z = make_uint4(0, 0, 0, 0);
        uint4* ho = (uint4*)Hout;
        for (int i = tid; i < (32 * HLD) / 8; i += TPB) ho[i] = z;
        unsigned* cm = (unsigned*)Cm;
        for (int i = tid; i < (PNODES * PNODES) / 2; i += TPB) cm[i] = 0u;
    }
    __syncthreads();

    // ---- Gram G = h @ h^T into padded 32x32 f32, unconditional D store
    {
        int rt = wave >> 1, nt = wave & 1;
        const _Float16* arow = Hin + (rt * 16 + lm) * HLD;
        const _Float16* brow = Hin + (nt * 16 + lm) * HLD;   // B[k][n] = h[n][k]
        v8f acc = zero8();
        for (int kt = 0; kt < kTiles; ++kt) {
            acc = wmma32(loadA16(arow, kt * 32, half), loadB16(brow, kt * 32, half), acc);
        }
        int n = nt * 16 + lm;
#pragma unroll
        for (int g = 0; g < 8; ++g) {
            int m = rt * 16 + g + 8 * half;
            GM[m * 32 + n] = acc[g];
        }
    }
    __syncthreads();

    // ---- top-K=5 neighbor selection per node p (branchless min-scans)
    if (tid < PNODES) {
        int p = tid;
        float Gpp = GM[p * 32 + p];
        float d2[PNODES];                       // registers (static indices only)
#pragma unroll
        for (int q = 0; q < PNODES; ++q)
            d2[q] = Gpp + GM[q * 32 + q] - 2.0f * GM[p * 32 + q];
        unsigned used = 1u << p;                // exclude self
#pragma unroll
        for (int j = 0; j < KNN; ++j) {
            float bd = 3.4e38f;
            int bq = 0;
#pragma unroll
            for (int q = 0; q < PNODES; ++q) {
                bool ok = (((used >> q) & 1u) == 0u) & (d2[q] < bd);
                bd = ok ? d2[q] : bd;           // v_cndmask, no branches
                bq = ok ? q : bq;
            }
            used |= 1u << bq;
            Cm[bq * PNODES + p] = (_Float16)1.0f;   // column p unique per lane
        }
    }
    __syncthreads();

    // ---- G is dead now: zero Mm (aliases GM) while computing degrees
    {
        uint4 z = make_uint4(0, 0, 0, 0);
        uint4* mm = (uint4*)Mm;
        for (int i = tid; i < (32 * 32) / 8; i += TPB) mm[i] = z;
    }
    if (tid < PNODES) {
        float s = 1.0f;
#pragma unroll
        for (int q = 0; q < PNODES; ++q) s += (float)Cm[tid * PNODES + q];
        dinvS[tid] = rsqrtf(s);
    }
    __syncthreads();

    // ---- M = dinv C dinv + diag(dinv^2), f16, zero-padded to 32x32
    if (tid < PNODES) {
        float dv = dinvS[tid];
#pragma unroll
        for (int q = 0; q < PNODES; ++q) {
            float m = dv * (float)Cm[tid * PNODES + q] * dinvS[q];
            if (q == tid) m += dv * dv;
            Mm[tid * 32 + q] = (_Float16)m;
        }
    }
    __syncthreads();

    // ---- t = h @ W^T, stored TRANSPOSED: tT[n][m] f16 (ld 32), one b128 store/lane
    for (int tile = wave; tile < 2 * nTiles; tile += 4) {
        int rt = tile & 1, nt = tile >> 1;
        const _Float16* arow = Hin + (rt * 16 + lm) * HLD;
        const _Float16* brow = Wf + (size_t)(nt * 16 + lm) * KpadW;   // B[k][n] = W[n][k]
        v8f acc = zero8();
        for (int kt = 0; kt < kTiles; ++kt) {
            if (kt + 1 < kTiles)
                __builtin_prefetch(brow + (kt + 1) * 32, 0, 1);
            acc = wmma32(loadA16(arow, kt * 32, half), loadB16(brow, kt * 32, half), acc);
        }
        v8h packed;
#pragma unroll
        for (int g = 0; g < 8; ++g) packed[g] = (_Float16)acc[g];
        int n = nt * 16 + lm;                       // D layout: lane n fixed, m = g + 8*half
        *(v8h*)(tT + n * 32 + rt * 16 + 8 * half) = packed;
    }
    __syncthreads();

    // ---- h_out = relu(M @ t + b): A = Mm (32x32, single K step), B = tT rows
    for (int tile = wave; tile < 2 * nTiles; tile += 4) {
        int rt = tile & 1, nt = tile >> 1;
        const _Float16* arow = Mm + (rt * 16 + lm) * 32;
        int n = nt * 16 + lm;
        const _Float16* brow = tT + n * 32;          // B[k][n] = tT[n][k]
        v8f acc = wmma32(loadA16(arow, 0, half), loadB16(brow, 0, half), zero8());
        float bb = (n < N_out) ? bvec[n] : 0.0f;
#pragma unroll
        for (int g = 0; g < 8; ++g) {
            int m = rt * 16 + g + 8 * half;
            float v = (m < PNODES && n < N_out) ? fmaxf(acc[g] + bb, 0.0f) : 0.0f;
            Hout[m * HLD + n] = (_Float16)v;        // unconditional, value-predicated
        }
    }
    __syncthreads();
}

// ------------------------------------------------------------- kernel 1 -----

__global__ __launch_bounds__(TPB) void gcn_kernel(
        const float* __restrict__ x,
        const _Float16* __restrict__ W1f, const float* __restrict__ b1,
        const _Float16* __restrict__ W2f, const float* __restrict__ b2,
        const _Float16* __restrict__ W3f, const float* __restrict__ b3,
        _Float16* __restrict__ h3out) {
    __shared__ alignas(16) _Float16 H0[32 * HLD];         // 20480 B
    __shared__ alignas(16) _Float16 H1[32 * HLD];         // 20480 B
    __shared__ alignas(16) _Float16 tT[304 * 32];         // 19456 B
    __shared__ alignas(16) float    GM[32 * 32];          //  4096 B (G f32 / M f16 union)
    __shared__ alignas(16) _Float16 Cm[PNODES * PNODES];  //   800 B
    __shared__ float dinvS[32];                           //   128 B  -> 65,440 total

    int tid = threadIdx.x;
    int lane = tid & 31;
    int wave = tid >> 5;
    size_t s = blockIdx.x;

    // zero H0, then load x[s] (20x128 f32) as f16 rows
    {
        uint4 z = make_uint4(0, 0, 0, 0);
        uint4* h0 = (uint4*)H0;
        for (int i = tid; i < (32 * HLD) / 8; i += TPB) h0[i] = z;
    }
    __syncthreads();
    {
        const v4f* xs = (const v4f*)(x + s * (size_t)(PNODES * 128));
        for (int i = tid; i < PNODES * 32; i += TPB) {   // 640 float4 chunks
            int vv = i >> 5, c4 = i & 31;
            v4f f = xs[i];
            v4h hchunk;
#pragma unroll
            for (int e = 0; e < 4; ++e) hchunk[e] = (_Float16)f[e];
            *(v4h*)(H0 + vv * HLD + c4 * 4) = hchunk;
        }
    }
    __syncthreads();

    //                in  out                       W    b  kT  nT Nout KpadW
    gcn_layer(H0, H1, tT, GM, Cm, dinvS, W1f, b1,  4, 19, 300, 128, tid, lane, wave);
    gcn_layer(H1, H0, tT, GM, Cm, dinvS, W2f, b2, 10,  7, 100, 320, tid, lane, wave);
    gcn_layer(H0, H1, tT, GM, Cm, dinvS, W3f, b3,  4,  4,  50, 128, tid, lane, wave);

    // flattened h3 (20*50 = 1000) zero-padded to 1024 f16 per sample
    _Float16* dst = h3out + s * 1024;
    for (int i = tid; i < 1024; i += TPB) {
        _Float16 v = (_Float16)0.0f;
        if (i < 1000) {
            int vv = i / 50, j = i - vv * 50;
            v = H1[vv * HLD + j];
        }
        dst[i] = v;
    }
}

// ------------------------------------------------------------- kernel 2 -----
// 16 samples / workgroup: relu((16x1024)@W4f^T + b4) -> FC5 -> softmax(3)

__global__ __launch_bounds__(TPB) void head_kernel(
        const _Float16* __restrict__ h3,
        const _Float16* __restrict__ W4f,    // 112 x 1024 f16, zero-padded
        const float* __restrict__ b4,
        const float* __restrict__ W5, const float* __restrict__ b5,
        float* __restrict__ out, int Bn) {
    __shared__ alignas(16) _Float16 Atile[16 * 1024];   // 32 KB
    __shared__ float T[16 * 112];
    __shared__ float logits[16 * 3];

    int tid = threadIdx.x;
    int lane = tid & 31;
    int wave = tid >> 5;
    int lm   = lane & 15;
    int half = (lane >> 4) & 1;
    size_t row0 = (size_t)blockIdx.x * 16;

    // stage 16x1024 f16 tile (uint4 chunks; rows past Bn zeroed)
    {
        uint4* at = (uint4*)Atile;
        const uint4* src = (const uint4*)(h3 + row0 * 1024);
        for (int i = tid; i < (16 * 1024) / 8; i += TPB) {
            size_t row = row0 + (size_t)(i >> 7);
            at[i] = (row < (size_t)Bn) ? src[i] : make_uint4(0, 0, 0, 0);
        }
    }
    __syncthreads();

    const int nTiles = 7;  // 100 -> 112
    for (int nt = wave; nt < nTiles; nt += 4) {
        const _Float16* arow = Atile + lm * 1024;
        __builtin_assume(arow != nullptr);
        int n = nt * 16 + lm;
        const _Float16* brow = W4f + (size_t)n * 1024;
        v8f acc = zero8();
        for (int kt = 0; kt < 32; ++kt) {
            if (kt + 1 < 32)
                __builtin_prefetch(brow + (kt + 1) * 32, 0, 1);
            acc = wmma32(loadA16(arow, kt * 32, half), loadB16(brow, kt * 32, half), acc);
        }
        float bb = (n < 100) ? b4[n] : 0.0f;
#pragma unroll
        for (int g = 0; g < 8; ++g) {
            int m = g + 8 * half;                  // 16 rows, all valid; n < 112 always
            T[m * 112 + n] = fmaxf(acc[g] + bb, 0.0f);
        }
    }
    __syncthreads();

    // FC5: 16 rows x 3 classes, dot length 100
    if (tid < 48) {
        int r = tid / 3, c = tid - 3 * (tid / 3);
        float sacc = b5[c];
        for (int k = 0; k < 100; ++k) sacc += T[r * 112 + k] * W5[c * 100 + k];
        logits[r * 3 + c] = sacc;
    }
    __syncthreads();

    if (tid < 16 && row0 + tid < (size_t)Bn) {
        float l0 = logits[tid * 3 + 0];
        float l1 = logits[tid * 3 + 1];
        float l2 = logits[tid * 3 + 2];
        float mx = fmaxf(l0, fmaxf(l1, l2));
        float e0 = __expf(l0 - mx), e1 = __expf(l1 - mx), e2 = __expf(l2 - mx);
        float inv = 1.0f / (e0 + e1 + e2);
        size_t o = (row0 + tid) * 3;
        out[o + 0] = e0 * inv;
        out[o + 1] = e1 * inv;
        out[o + 2] = e2 * inv;
    }
}

// -------------------------------------------------- weight pre-conversion ---

__global__ void convert_w(const float* __restrict__ src, _Float16* __restrict__ dst,
                          int Nout, int Nin, int Npad, int Kpad) {
    int i = blockIdx.x * blockDim.x + threadIdx.x;
    int total = Npad * Kpad;
    if (i >= total) return;
    int n = i / Kpad, k = i - n * Kpad;
    float v = (n < Nout && k < Nin) ? src[n * Nin + k] : 0.0f;
    dst[i] = (_Float16)v;
}

// --------------------------------------------------------------- launch -----

extern "C" void kernel_launch(void* const* d_in, const int* in_sizes, int n_in,
                              void* d_out, int out_size, void* d_ws, size_t ws_size,
                              hipStream_t stream) {
    const float* x  = (const float*)d_in[0];
    const float* W1 = (const float*)d_in[1];
    const float* b1 = (const float*)d_in[2];
    const float* W2 = (const float*)d_in[3];
    const float* b2 = (const float*)d_in[4];
    const float* W3 = (const float*)d_in[5];
    const float* b3 = (const float*)d_in[6];
    const float* W4 = (const float*)d_in[7];
    const float* b4 = (const float*)d_in[8];
    const float* W5 = (const float*)d_in[9];
    const float* b5 = (const float*)d_in[10];
    float* outp = (float*)d_out;

    int Bn = in_sizes[0] / (PNODES * 128);

    // workspace layout (f16 units)
    _Float16* ws   = (_Float16*)d_ws;
    _Float16* h3   = ws;                         size_t ofs = (size_t)Bn * 1024;
    _Float16* W1f  = ws + ofs;                   ofs += 304 * 128;   // 19 tiles x Kpad 128
    _Float16* W2f  = ws + ofs;                   ofs += 112 * 320;   //  7 tiles x Kpad 320
    _Float16* W3f  = ws + ofs;                   ofs += 64 * 128;    //  4 tiles x Kpad 128
    _Float16* W4f  = ws + ofs;                   ofs += 112 * 1024;  //  7 tiles x Kpad 1024

    convert_w<<<(304 * 128 + 255) / 256, 256, 0, stream>>>(W1, W1f, 300, 128, 304, 128);
    convert_w<<<(112 * 320 + 255) / 256, 256, 0, stream>>>(W2, W2f, 100, 300, 112, 320);
    convert_w<<<(64 * 128 + 255) / 256, 256, 0, stream>>>(W3, W3f, 50, 100, 64, 128);
    convert_w<<<(112 * 1024 + 255) / 256, 256, 0, stream>>>(W4, W4f, 100, 1000, 112, 1024);

    gcn_kernel<<<Bn, TPB, 0, stream>>>(x, W1f, b1, W2f, b2, W3f, b3, h3);
    head_kernel<<<(Bn + 15) / 16, TPB, 0, stream>>>(h3, W4f, b4, W5, b5, outp, Bn);
}
